// SMARTAgentDecoder_44676249813399
// MI455X (gfx1250) — matrix operand order repaired and checked by reading, not compiled, not run on hardware
//
#include <hip/hip_runtime.h>
#include <math.h>

// ============================================================================
// SMART agent decoder forward for gfx1250 (MI455X).
//   - all dense projections via v_wmma_f32_16x16x32_f16 (f32 accum)
//   - weights pre-converted to f16 and pre-transposed ([n][k], K padded to 32)
//     once per launch
//   - B tiles double-buffered via CDNA5 async copies
//     (global_load_async_to_lds_b128 + s_wait_asynccnt), overlapping the
//     weight fetch with A staging and WMMA compute
//   - activation tiles staged branch-free; tail path pipelines 8 loads
//   - irregular graph ops (edge softmax / segment sums) via atomic kernels
//
// Input pointer order = jax tree_flatten of setup_inputs() (dict keys sorted
// alphabetically, depth-first).  156 leaves total; indices below.
// ============================================================================

typedef __attribute__((ext_vector_type(16))) _Float16 v16h;
typedef __attribute__((ext_vector_type(8)))  _Float16 v8h;
typedef __attribute__((ext_vector_type(8)))  float    v8f;

// ---------------- problem constants ----------------
constexpr int NA  = 1024;
constexpr int NS  = 18;
constexpr int NT  = NA * NS;       // 18432 tokens
constexpr int DD  = 128;
constexpr int NH  = 8;
constexpr int FF  = 64;
constexpr int FT_W = 2 * FF + 1;   // 129
constexpr int FT_LD = 132;         // padded for vector path
constexpr int NLAYER = 3;
constexpr int VV  = 512;
constexpr float PI_F  = 3.14159265358979323846f;
constexpr float TPI_F = 6.28318530717958647692f;

// ---------------- input leaf indices (jax tree_flatten order) ----------------
enum {
  IN_A2A_DST = 0, IN_A2A_SRC = 1, IN_AGENT_SHAPE = 2, IN_AGENT_TYPE = 3,
  IN_HEAD_A = 4,
  B_AATTN = 5, B_FUSION = 29, B_HEAD = 35, B_PTATTN = 41,
  B_RA = 65, B_RPT = 76, B_RT = 87, B_SHAPE = 98, B_TATTN = 104,
  B_TOKEMB = 128, IN_TYPE_EMB = 134, B_XA = 135,
  IN_POS_A = 146, IN_PT2A_DST = 147, IN_PT2A_SRC = 148, IN_PT_ORIENT = 149,
  IN_PT_POS = 150, IN_T_DST = 151, IN_T_SRC = 152, IN_TOKEN_IDX = 153,
  IN_TRAJ = 154, IN_X_PT = 155
};
enum {
  AT_Wf1 = 0, AT_Wf2 = 1, AT_Wg = 2, AT_Wk = 3, AT_Wkr = 4, AT_Wo = 5,
  AT_Wq = 6, AT_Ws = 7, AT_Wv = 8, AT_Wvr = 9, AT_bf1 = 10, AT_bf2 = 11,
  AT_bg = 12, AT_bo = 13, AT_bq = 14, AT_bs = 15, AT_ln2_b = 16, AT_ln2_g = 17,
  AT_lndst_b = 18, AT_lndst_g = 19, AT_lnr_b = 20, AT_lnr_g = 21,
  AT_lnsrc_b = 22, AT_lnsrc_g = 23
};
enum { ML_W1 = 0, ML_W2 = 1, ML_b = 2, ML_b1 = 3, ML_b2 = 4, ML_g = 5 };
enum { FO_W1 = 0, FO_W2 = 1, FO_b1 = 2, FO_b2 = 3, FO_bln = 4, FO_freqs = 5,
       FO_g1 = 6, FO_oW = 7, FO_ob = 8, FO_obn = 9, FO_og = 10 };

// ============================================================================
// weight prep: W[s][K][N] (f32) -> out[s][N][ldk] (f16, transposed, k-padded)
// ============================================================================
__global__ void k_wprep(const float* __restrict__ W, _Float16* __restrict__ out,
                        int K, int N, int ldk, long total)
{
  long i = (long)blockIdx.x * 256 + threadIdx.x;
  if (i >= total) return;
  int k = (int)(i % ldk);
  long t = i / ldk;
  int n = (int)(t % N);
  int s = (int)(t / N);
  float v = (k < K) ? W[((size_t)s * K + k) * N + n] : 0.f;
  out[i] = (_Float16)v;
}

// ============================================================================
// CDNA5 async copy: 16 bytes global -> LDS, tracked by ASYNCcnt.
// lds_addr = wave-relative LDS byte address (low 32 bits of generic pointer).
// ============================================================================
__device__ __forceinline__ void async_b128(unsigned lds_addr, const void* g)
{
  asm volatile("global_load_async_to_lds_b128 %0, %1, off"
               :: "v"(lds_addr), "v"(g) : "memory");
}

// ============================================================================
// WMMA GEMM:  C[M,N] = act( A[M,K](f32) @ Bt[N,K](f16,transposed) + bias (+C) )
// 256 threads = 8 waves; tile BM=64 x BN=128; waves 2x4, each 32x32 (2x2 frags)
// Requires: N multiple of 128 (all GEMMs here are N=128 or 512), ldk = K
// rounded up to 32 with zero padding.
// B tile double-buffered via async global->LDS copies.
// ============================================================================
#define BM 64
#define BN 128
#define BK 32

__global__ __launch_bounds__(256)
void k_gemm(const float* __restrict__ A, int lda,
            const _Float16* __restrict__ Bt, int ldk,
            const float* __restrict__ bias,
            float* __restrict__ C, int ldc,
            int M, int N, int K, int addC, int act)
{
  __shared__ __align__(16) _Float16 sA[BM * BK];      // [m][k]
  __shared__ __align__(16) _Float16 sB[2][BN * BK];   // [n][k], ping-pong

  const int tid  = threadIdx.x;
  const int wave = tid >> 5;
  const int lane = tid & 31;
  const int m0 = blockIdx.x * BM;
  const int n0 = blockIdx.y * BN;
  const int wm = (wave & 1) * 32;
  const int wn = (wave >> 1) * 32;
  const int g   = lane >> 4;
  const int l16 = lane & 15;

  // per-thread B-tile transfer slot: 32 bytes = 2 async b128 ops per tile
  const int nn = tid >> 1, kc = (tid & 1) * 16;
  const _Float16* brow = Bt + (size_t)(n0 + nn) * ldk + kc;

  v8f acc[2][2] = {};

  // prologue: kick off B tile 0 (2 outstanding async ops per wave-lane-slot)
  {
    unsigned la = (unsigned)(uintptr_t)&sB[0][nn * BK + kc];
    async_b128(la, brow);
    async_b128(la + 16, brow + 8);
  }

  int cur = 0;
  for (int k0 = 0; k0 < K; k0 += BK) {
    // ---- stage A: f32 -> f16 (sync, through VGPRs) ----
    if (m0 + BM <= M && k0 + BK <= K && (lda & 3) == 0) {
      int mm = tid >> 2, kk = (tid & 3) * 8;
      const float* ap = A + (size_t)(m0 + mm) * lda + k0 + kk;
      float4 f0 = *(const float4*)ap;
      float4 f1 = *(const float4*)(ap + 4);
      v8h h;
      h[0] = (_Float16)f0.x; h[1] = (_Float16)f0.y;
      h[2] = (_Float16)f0.z; h[3] = (_Float16)f0.w;
      h[4] = (_Float16)f1.x; h[5] = (_Float16)f1.y;
      h[6] = (_Float16)f1.z; h[7] = (_Float16)f1.w;
      *(v8h*)&sA[mm * BK + kk] = h;
    } else {
      // tail path: 8 independent clamped loads, then masked stores
      float va[8]; bool oka[8];
#pragma unroll
      for (int u = 0; u < 8; ++u) {
        int i = tid + u * 256;
        int mm = i >> 5, kk = i & 31;
        int gm = m0 + mm, gk = k0 + kk;
        bool ok = (gm < M) && (gk < K);
        oka[u] = ok;
        va[u] = A[ok ? ((size_t)gm * lda + gk) : 0];
      }
#pragma unroll
      for (int u = 0; u < 8; ++u)
        sA[tid + u * 256] = (_Float16)(oka[u] ? va[u] : 0.f);
    }
    // ---- B pipeline: issue next tile, wait for current ----
    bool more = (k0 + BK) < K;
    if (more) {
      unsigned la = (unsigned)(uintptr_t)&sB[cur ^ 1][nn * BK + kc];
      async_b128(la, brow + (k0 + BK));
      async_b128(la + 16, brow + (k0 + BK) + 8);
      asm volatile("s_wait_asynccnt 0x2" ::: "memory");  // current tile landed
    } else {
      asm volatile("s_wait_asynccnt 0x0" ::: "memory");
    }
    __syncthreads();

    // ---- fragments (ISA 7.12.2 layouts) ----
    v16h af[2], bf[2];
    for (int i = 0; i < 2; ++i) {
      int row = wm + i * 16 + l16;
      const v8h* p = (const v8h*)&sA[row * BK + g * 8];
      ((v8h*)&af[i])[0] = p[0];
      ((v8h*)&af[i])[1] = p[2];     // +16 halves
    }
    for (int j = 0; j < 2; ++j) {
      int col = wn + j * 16 + l16;
      const v8h* p = (const v8h*)&sB[cur][col * BK + g * 8];
      ((v8h*)&bf[j])[0] = p[0];
      ((v8h*)&bf[j])[1] = p[2];
    }
    for (int i = 0; i < 2; ++i)
      for (int j = 0; j < 2; ++j)
        acc[i][j] = __builtin_amdgcn_wmma_f32_16x16x32_f16(
            false, af[i], false, bf[j], (short)0, acc[i][j], false, false);
    __syncthreads();
    cur ^= 1;
  }

  // ---- epilogue: elem r -> m = r + g*8, n = lane%16 ----
  for (int i = 0; i < 2; ++i)
    for (int j = 0; j < 2; ++j)
      for (int r = 0; r < 8; ++r) {
        int m = m0 + wm + i * 16 + g * 8 + r;
        int n = n0 + wn + j * 16 + l16;
        if (m < M && n < N) {
          float v = acc[i][j][r];
          if (bias) v += bias[n];
          size_t off = (size_t)m * ldc + n;
          if (addC) v += C[off];
          if (act == 1)      v = v > 0.f ? v : 0.f;
          else if (act == 2) v = 1.f / (1.f + __expf(-v));
          C[off] = v;
        }
      }
}

// ============================================================================
// LayerNorm over width 128, one wave32 per row.  Optional fused ReLU.
// ============================================================================
__global__ __launch_bounds__(128)
void k_ln(const float* __restrict__ X, int ldx,
          const float* __restrict__ gam, const float* __restrict__ bet,
          float* __restrict__ Y, int ldy, int M, int relu)
{
  int row  = blockIdx.x * 4 + (threadIdx.x >> 5);
  int lane = threadIdx.x & 31;
  if (row >= M) return;
  const float* x = X + (size_t)row * ldx;
  float v[4]; float s = 0.f;
  for (int i = 0; i < 4; ++i) { v[i] = x[lane + 32 * i]; s += v[i]; }
  for (int o = 16; o >= 1; o >>= 1) s += __shfl_xor(s, o, 32);
  float mean = s * (1.f / 128.f);
  float var = 0.f;
  for (int i = 0; i < 4; ++i) { float d = v[i] - mean; var += d * d; }
  for (int o = 16; o >= 1; o >>= 1) var += __shfl_xor(var, o, 32);
  float rstd = rsqrtf(var * (1.f / 128.f) + 1e-5f);
  float* y = Y + (size_t)row * ldy;
  for (int i = 0; i < 4; ++i) {
    int c = lane + 32 * i;
    float o = (v[i] - mean) * rstd * gam[c] + bet[c];
    if (relu) o = fmaxf(o, 0.f);
    y[c] = o;
  }
}

// ============================================================================
// elementwise / graph kernels
// ============================================================================
__device__ inline float wrapf(float a) {
  float x = fmodf(a + PI_F, TPI_F);
  if (x < 0.f) x += TPI_F;
  return x - PI_F;
}
__device__ inline void atomicMaxF(float* addr, float val) {
  int* ai = (int*)addr;
  int old = __float_as_int(*addr);
  while (val > __int_as_float(old)) {
    int prev = atomicCAS(ai, old, __float_as_int(val));
    if (prev == old) break;
    old = prev;
  }
}

__global__ void k_fill(float* p, long n, float v) {
  long i = (long)blockIdx.x * 256 + threadIdx.x;
  if (i < n) p[i] = v;
}

__global__ void k_tokfeat(const float* __restrict__ pos,
                          const float* __restrict__ head,
                          float* __restrict__ hvf, float* __restrict__ feat2)
{
  int n = blockIdx.x * 256 + threadIdx.x;
  if (n >= NT) return;
  int s = n % NS;
  float h = head[n];
  float cx = cosf(h), cy = sinf(h);
  hvf[2 * n] = cx; hvf[2 * n + 1] = cy;
  float mx = 0.f, my = 0.f;
  if (s > 0) {
    mx = pos[2 * n] - pos[2 * (n - 1)];
    my = pos[2 * n + 1] - pos[2 * (n - 1) + 1];
  }
  feat2[2 * n]     = sqrtf(mx * mx + my * my);
  feat2[2 * n + 1] = atan2f(cx * my - cy * mx, cx * mx + cy * my);
}

__global__ void k_gather_tok(const float* __restrict__ tables,
                             const int* __restrict__ atype,
                             const int* __restrict__ tokidx,
                             float* __restrict__ xcat)
{
  long i = (long)blockIdx.x * 256 + threadIdx.x;
  if (i >= (long)NT * DD) return;
  int n = (int)(i >> 7), c = (int)(i & 127);
  int a = n / NS;
  xcat[(size_t)n * 256 + c] =
      tables[((size_t)atype[a] * VV + tokidx[n]) * DD + c];
}

__global__ void k_cat(const float* __restrict__ type_emb,
                      const int* __restrict__ atype,
                      const float* __restrict__ shp, float* __restrict__ cat)
{
  long i = (long)blockIdx.x * 256 + threadIdx.x;
  if (i >= (long)NA * DD) return;
  int a = (int)(i >> 7), c = (int)(i & 127);
  cat[i] = type_emb[(size_t)atype[a] * DD + c] + shp[i];
}

__global__ void k_efeat_t(const float* __restrict__ pos,
                          const float* __restrict__ head,
                          const float* __restrict__ hvf,
                          const int* __restrict__ src, const int* __restrict__ dst,
                          float* __restrict__ out, int E)
{
  int e = blockIdx.x * 256 + threadIdx.x;
  if (e >= E) return;
  int s = src[e], d = dst[e];
  float rx = pos[2 * s] - pos[2 * d], ry = pos[2 * s + 1] - pos[2 * d + 1];
  float cx = hvf[2 * d], cy = hvf[2 * d + 1];
  out[e * 4 + 0] = sqrtf(rx * rx + ry * ry);
  out[e * 4 + 1] = atan2f(cx * ry - cy * rx, cx * rx + cy * ry);
  out[e * 4 + 2] = wrapf(head[s] - head[d]);
  out[e * 4 + 3] = (float)((d % NS) - (s % NS));
}

__global__ void k_efeat_pt(const float* __restrict__ ppos,
                           const float* __restrict__ porient,
                           const float* __restrict__ pos,
                           const float* __restrict__ head,
                           const float* __restrict__ hvf,
                           const int* __restrict__ src, const int* __restrict__ dst,
                           float* __restrict__ out, int E)
{
  int e = blockIdx.x * 256 + threadIdx.x;
  if (e >= E) return;
  int s = src[e], d = dst[e];
  float rx = ppos[2 * s] - pos[2 * d], ry = ppos[2 * s + 1] - pos[2 * d + 1];
  float cx = hvf[2 * d], cy = hvf[2 * d + 1];
  out[e * 3 + 0] = sqrtf(rx * rx + ry * ry);
  out[e * 3 + 1] = atan2f(cx * ry - cy * rx, cx * rx + cy * ry);
  out[e * 3 + 2] = wrapf(porient[s] - head[d]);
}

__global__ void k_efeat_a(const float* __restrict__ pos,
                          const float* __restrict__ head,
                          const float* __restrict__ hvf,
                          const int* __restrict__ src, const int* __restrict__ dst,
                          float* __restrict__ out, int E)
{
  int e = blockIdx.x * 256 + threadIdx.x;
  if (e >= E) return;
  int s = src[e], d = dst[e];
  float rx = pos[2 * s] - pos[2 * d], ry = pos[2 * s + 1] - pos[2 * d + 1];
  float cx = hvf[2 * d], cy = hvf[2 * d + 1];
  out[e * 3 + 0] = sqrtf(rx * rx + ry * ry);
  out[e * 3 + 1] = atan2f(cx * ry - cy * rx, cx * rx + cy * ry);
  out[e * 3 + 2] = wrapf(head[s] - head[d]);
}

__global__ void k_ft(const float* __restrict__ X, int din, int i,
                     const float* __restrict__ freqs,
                     float* __restrict__ ft, int M)
{
  long idx = (long)blockIdx.x * 256 + threadIdx.x;
  if (idx >= (long)M * FT_W) return;
  int m = (int)(idx / FT_W), j = (int)(idx % FT_W);
  float x = X[(size_t)m * din + i];
  float v;
  if (j < FF)        v = cosf(x * freqs[i * FF + j] * TPI_F);
  else if (j < 2*FF) v = sinf(x * freqs[i * FF + (j - FF)] * TPI_F);
  else               v = x;
  ft[(size_t)m * FT_LD + j] = v;
}

__global__ void k_addcat(float* __restrict__ h, const float* __restrict__ cat,
                         int catDiv, int M)
{
  long i = (long)blockIdx.x * 256 + threadIdx.x;
  if (i >= (long)M * DD) return;
  int m = (int)(i >> 7), c = (int)(i & 127);
  h[i] += cat[(size_t)(m / catDiv) * DD + c];
}

__global__ void k_sim(const float* __restrict__ q, const float* __restrict__ kb,
                      const float* __restrict__ ke,
                      const int* __restrict__ src, const int* __restrict__ dst,
                      float* __restrict__ sim, int E)
{
  int i = blockIdx.x * 256 + threadIdx.x;
  if (i >= E * NH) return;
  int e = i >> 3, h = i & 7;
  const float* qp = q  + (size_t)dst[e] * DD + h * 16;
  const float* kp = kb + (size_t)src[e] * DD + h * 16;
  const float* ap = ke + (size_t)e * DD + h * 16;
  float s = 0.f;
  for (int d = 0; d < 16; ++d) s += qp[d] * (kp[d] + ap[d]);
  sim[i] = s * 0.25f;   // 1/sqrt(16)
}

__global__ void k_amax(const float* __restrict__ sim, const int* __restrict__ dst,
                       float* __restrict__ m, int E)
{
  int i = blockIdx.x * 256 + threadIdx.x;
  if (i >= E * NH) return;
  int e = i >> 3, h = i & 7;
  atomicMaxF(&m[(size_t)dst[e] * NH + h], sim[i]);
}

__global__ void k_aexp(float* __restrict__ sim, const int* __restrict__ dst,
                       const float* __restrict__ m, float* __restrict__ den, int E)
{
  int i = blockIdx.x * 256 + threadIdx.x;
  if (i >= E * NH) return;
  int e = i >> 3, h = i & 7;
  float v = __expf(sim[i] - m[(size_t)dst[e] * NH + h]);
  sim[i] = v;
  atomicAdd(&den[(size_t)dst[e] * NH + h], v);
}

__global__ void k_anorm(float* __restrict__ sim, const int* __restrict__ dst,
                        const float* __restrict__ den, int E)
{
  int i = blockIdx.x * 256 + threadIdx.x;
  if (i >= E * NH) return;
  int e = i >> 3, h = i & 7;
  sim[i] = sim[i] / (den[(size_t)dst[e] * NH + h] + 1e-9f);
}

// msg accumulates into a strided buffer (cat2[:, :128], ldm = 256)
__global__ void k_scatter(const float* __restrict__ vb, const float* __restrict__ ve,
                          const float* __restrict__ a,
                          const int* __restrict__ src, const int* __restrict__ dst,
                          float* __restrict__ msg, int ldm, int E)
{
  long i = (long)blockIdx.x * 256 + threadIdx.x;
  if (i >= (long)E * DD) return;
  int e = (int)(i >> 7), c = (int)(i & 127);
  float val = (vb[(size_t)src[e] * DD + c] + ve[i]) * a[(size_t)e * NH + (c >> 4)];
  atomicAdd(&msg[(size_t)dst[e] * ldm + c], val);
}

__global__ void k_mix(const float* __restrict__ msg, int ldm,
                      const float* __restrict__ g, float* __restrict__ s, int M)
{
  long i = (long)blockIdx.x * 256 + threadIdx.x;
  if (i >= (long)M * DD) return;
  int m = (int)(i >> 7), c = (int)(i & 127);
  float mg = msg[(size_t)m * ldm + c];
  s[i] = mg + g[i] * (s[i] - mg);
}

// ============================================================================
// host-side helpers
// ============================================================================
struct PW { const _Float16* p; int ldk; size_t slice; };

static inline const float* lf(void* const* din, int idx) { return (const float*)din[idx]; }
static inline const int*   li(void* const* din, int idx) { return (const int*)din[idx]; }

static inline void gemm(hipStream_t st, const float* A, int lda,
                        const PW& pw, int slice, const float* bias,
                        float* C, int ldc, int M, int N, int K,
                        int addC, int act)
{
  dim3 g((M + BM - 1) / BM, (N + BN - 1) / BN);
  k_gemm<<<g, 256, 0, st>>>(A, lda, pw.p + (size_t)slice * pw.slice, pw.ldk,
                            bias, C, ldc, M, N, K, addC, act);
}

static inline void ln_rows(hipStream_t st, const float* X, int ldx,
                           const float* g, const float* b,
                           float* Y, int ldy, int M, int relu)
{
  k_ln<<<(M + 3) / 4, 128, 0, st>>>(X, ldx, g, b, Y, ldy, M, relu);
}

static inline void fill(hipStream_t st, float* p, long n, float v)
{
  k_fill<<<(unsigned)((n + 255) / 256), 256, 0, st>>>(p, n, v);
}

struct MlpPW  { PW w1, w2; };
struct FourPW { PW w1, w2, ow; };
struct AttnPW { PW q, k, v, kr, vr, s, g, o, f1, f2; };

// relu(LN(X@W1+b1)) @ W2 + b2   (leaf biases come from d_in, slice-indexed)
static void mlp_apply(hipStream_t st, const float* X, int lda, int M, int din,
                      int dout, const MlpPW& mw, int slice,
                      void* const* dinp, int base,
                      float* out, int ldout, float* scratch)
{
  gemm(st, X, lda, mw.w1, slice, lf(dinp, base + ML_b1) + (size_t)slice * DD,
       scratch, DD, M, DD, din, 0, 0);
  ln_rows(st, scratch, DD, lf(dinp, base + ML_g) + (size_t)slice * DD,
          lf(dinp, base + ML_b) + (size_t)slice * DD, scratch, DD, M, 1);
  gemm(st, scratch, DD, mw.w2, slice, lf(dinp, base + ML_b2) + (size_t)slice * dout,
       out, ldout, M, dout, DD, 0, 0);
}

static void fourier_apply(hipStream_t st, const float* X, int M, int din,
                          const FourPW& fw, void* const* dinp, int base,
                          const float* cat, int catDiv,
                          float* out, int ldout, float* scratch)
{
  const float* b1  = lf(dinp, base + FO_b1);
  const float* b2  = lf(dinp, base + FO_b2);
  const float* bln = lf(dinp, base + FO_bln);
  const float* freqs = lf(dinp, base + FO_freqs);
  const float* g1  = lf(dinp, base + FO_g1);
  const float* ob  = lf(dinp, base + FO_ob);
  const float* obn = lf(dinp, base + FO_obn);
  const float* og  = lf(dinp, base + FO_og);

  float* ft   = scratch;
  float* hbuf = ft + (size_t)M * FT_LD;
  float* hsum = hbuf + (size_t)M * DD;

  for (int i = 0; i < din; ++i) {
    long n = (long)M * FT_W;
    k_ft<<<(unsigned)((n + 255) / 256), 256, 0, st>>>(X, din, i, freqs, ft, M);
    gemm(st, ft, FT_LD, fw.w1, i, b1 + (size_t)i * DD, hbuf, DD, M, DD, FT_W, 0, 0);
    ln_rows(st, hbuf, DD, g1 + (size_t)i * DD, bln + (size_t)i * DD, hbuf, DD, M, 1);
    gemm(st, hbuf, DD, fw.w2, i, b2 + (size_t)i * DD, hsum, DD, M, DD, DD,
         (i > 0) ? 1 : 0, 0);
  }
  if (cat) {
    long n = (long)M * DD;
    k_addcat<<<(unsigned)((n + 255) / 256), 256, 0, st>>>(hsum, cat, catDiv, M);
  }
  ln_rows(st, hsum, DD, og, obn, hbuf, DD, M, 1);
  gemm(st, hbuf, DD, fw.ow, 0, ob, out, ldout, M, DD, DD, 0, 0);
}

struct AttnWS {
  float *xs, *xd, *q, *kb, *vb, *rr, *ke, *ve, *sim, *mb, *den, *cat2,
        *gb, *sb, *hbig;
};

static void attn_apply(hipStream_t st, void* const* din, int base, int layer,
                       const AttnPW& pw,
                       const float* xsrc, int Msrc, float* feat,
                       const float* r, const int* src, const int* dst, int E,
                       const AttnWS& w)
{
  const float* bf1 = lf(din, base + AT_bf1) + (size_t)layer * 4 * DD;
  const float* bf2 = lf(din, base + AT_bf2) + (size_t)layer * DD;
  const float* bg  = lf(din, base + AT_bg)  + (size_t)layer * DD;
  const float* bo  = lf(din, base + AT_bo)  + (size_t)layer * DD;
  const float* bq  = lf(din, base + AT_bq)  + (size_t)layer * DD;
  const float* bs  = lf(din, base + AT_bs)  + (size_t)layer * DD;
  const float* l2b = lf(din, base + AT_ln2_b) + (size_t)layer * DD;
  const float* l2g = lf(din, base + AT_ln2_g) + (size_t)layer * DD;
  const float* ldb_ = lf(din, base + AT_lndst_b) + (size_t)layer * DD;
  const float* ldg_ = lf(din, base + AT_lndst_g) + (size_t)layer * DD;
  const float* lrb = lf(din, base + AT_lnr_b) + (size_t)layer * DD;
  const float* lrg = lf(din, base + AT_lnr_g) + (size_t)layer * DD;
  const float* lsb = lf(din, base + AT_lnsrc_b) + (size_t)layer * DD;
  const float* lsg = lf(din, base + AT_lnsrc_g) + (size_t)layer * DD;

  float* xd = w.cat2 + DD;    // LN(x_dst) lives in cat2[:,128:], ld = 256

  // 1) layer norms (zero cat2 first: msg accumulator occupies cat2[:,:128])
  ln_rows(st, xsrc, DD, lsg, lsb, w.xs, DD, Msrc, 0);
  fill(st, w.cat2, (long)NT * 2 * DD, 0.f);
  ln_rows(st, feat, DD, ldg_, ldb_, xd, 2 * DD, NT, 0);
  ln_rows(st, r, DD, lrg, lrb, w.rr, DD, E, 0);
  // 2) projections (WMMA)
  gemm(st, xd, 2 * DD, pw.q, layer, bq,     w.q,  DD, NT,   DD, DD, 0, 0);
  gemm(st, w.xs, DD, pw.k, layer, nullptr,  w.kb, DD, Msrc, DD, DD, 0, 0);
  gemm(st, w.xs, DD, pw.v, layer, nullptr,  w.vb, DD, Msrc, DD, DD, 0, 0);
  gemm(st, w.rr, DD, pw.kr, layer, nullptr, w.ke, DD, E,    DD, DD, 0, 0);
  gemm(st, w.rr, DD, pw.vr, layer, nullptr, w.ve, DD, E,    DD, DD, 0, 0);
  // 3) edge softmax + message aggregation
  fill(st, w.mb,  (long)NT * NH, -1e30f);
  fill(st, w.den, (long)NT * NH, 0.f);
  int nsim = E * NH;
  k_sim<<<(nsim + 255) / 256, 256, 0, st>>>(w.q, w.kb, w.ke, src, dst, w.sim, E);
  k_amax<<<(nsim + 255) / 256, 256, 0, st>>>(w.sim, dst, w.mb, E);
  k_aexp<<<(nsim + 255) / 256, 256, 0, st>>>(w.sim, dst, w.mb, w.den, E);
  k_anorm<<<(nsim + 255) / 256, 256, 0, st>>>(w.sim, dst, w.den, E);
  long nsc = (long)E * DD;
  k_scatter<<<(unsigned)((nsc + 255) / 256), 256, 0, st>>>(
      w.vb, w.ve, w.sim, src, dst, w.cat2, 2 * DD, E);
  // 4) gate + output projection (+residual into feat)
  gemm(st, w.cat2, 2 * DD, pw.g, layer, bg, w.gb, DD, NT, DD, 2 * DD, 0, 2);
  gemm(st, xd, 2 * DD, pw.s, layer, bs, w.sb, DD, NT, DD, DD, 0, 0);
  k_mix<<<(unsigned)(((long)NT * DD + 255) / 256), 256, 0, st>>>(
      w.cat2, 2 * DD, w.gb, w.sb, NT);
  gemm(st, w.sb, DD, pw.o, layer, bo, feat, DD, NT, DD, DD, 1, 0);
  // 5) FFN (+residual into feat)
  ln_rows(st, feat, DD, l2g, l2b, w.xd, DD, NT, 0);
  gemm(st, w.xd, DD, pw.f1, layer, bf1, w.hbig, 4 * DD, NT, 4 * DD, DD, 0, 1);
  gemm(st, w.hbig, 4 * DD, pw.f2, layer, bf2, feat, DD, NT, DD, 4 * DD, 1, 0);
}

// ============================================================================
// entry point
// ============================================================================
extern "C" void kernel_launch(void* const* d_in, const int* in_sizes, int n_in,
                              void* d_out, int out_size, void* d_ws, size_t ws_size,
                              hipStream_t stream)
{
  (void)n_in; (void)out_size; (void)ws_size;
  hipStream_t st = stream;

  const int Et = in_sizes[IN_T_SRC];
  const int Ep = in_sizes[IN_PT2A_SRC];
  const int Ea = in_sizes[IN_A2A_SRC];
  const int Ppt = in_sizes[IN_X_PT] / DD;
  int Emax = Et > Ep ? Et : Ep; if (Ea > Emax) Emax = Ea;
  const int MS = (Ppt > NT) ? Ppt : NT;

  // ---------------- weight prep region: first 32 MB of ws ----------------
  _Float16* wbase = (_Float16*)d_ws;
  size_t woff = 0;
  auto prep = [&](const float* W, int K, int N, int ns) -> PW {
    int ldk = (K + 31) & ~31;
    PW pw{wbase + woff, ldk, (size_t)N * ldk};
    long total = (long)ns * N * ldk;
    k_wprep<<<(unsigned)((total + 255) / 256), 256, 0, st>>>(
        W, wbase + woff, K, N, ldk, total);
    woff += (size_t)total;
    return pw;
  };
  auto prepAttn = [&](int base) -> AttnPW {
    AttnPW p;
    p.f1 = prep(lf(d_in, base + AT_Wf1), DD, 4 * DD, NLAYER);
    p.f2 = prep(lf(d_in, base + AT_Wf2), 4 * DD, DD, NLAYER);
    p.g  = prep(lf(d_in, base + AT_Wg),  2 * DD, DD, NLAYER);
    p.k  = prep(lf(d_in, base + AT_Wk),  DD, DD, NLAYER);
    p.kr = prep(lf(d_in, base + AT_Wkr), DD, DD, NLAYER);
    p.o  = prep(lf(d_in, base + AT_Wo),  DD, DD, NLAYER);
    p.q  = prep(lf(d_in, base + AT_Wq),  DD, DD, NLAYER);
    p.s  = prep(lf(d_in, base + AT_Ws),  DD, DD, NLAYER);
    p.v  = prep(lf(d_in, base + AT_Wv),  DD, DD, NLAYER);
    p.vr = prep(lf(d_in, base + AT_Wvr), DD, DD, NLAYER);
    return p;
  };
  auto prepFour = [&](int base, int din) -> FourPW {
    FourPW f;
    f.w1 = prep(lf(d_in, base + FO_W1), FT_W, DD, din);
    f.w2 = prep(lf(d_in, base + FO_W2), DD, DD, din);
    f.ow = prep(lf(d_in, base + FO_oW), DD, DD, 1);
    return f;
  };
  auto prepMlp = [&](int base, int din, int dout, int ns) -> MlpPW {
    MlpPW m;
    m.w1 = prep(lf(d_in, base + ML_W1), din, DD, ns);
    m.w2 = prep(lf(d_in, base + ML_W2), DD, dout, ns);
    return m;
  };

  AttnPW pw_t  = prepAttn(B_TATTN);
  AttnPW pw_pt = prepAttn(B_PTATTN);
  AttnPW pw_a  = prepAttn(B_AATTN);
  FourPW fw_xa = prepFour(B_XA, 2);
  FourPW fw_rt = prepFour(B_RT, 4);
  FourPW fw_rp = prepFour(B_RPT, 3);
  FourPW fw_ra = prepFour(B_RA, 3);
  MlpPW mw_tok  = prepMlp(B_TOKEMB, 8, DD, 3);
  MlpPW mw_shp  = prepMlp(B_SHAPE, 3, DD, 1);
  MlpPW mw_fus  = prepMlp(B_FUSION, 2 * DD, DD, 1);
  MlpPW mw_head = prepMlp(B_HEAD, DD, VV, 1);

  // ---------------- float workspace after the 32 MB weight region ----------
  float* w = (float*)((char*)d_ws + (32u << 20));
  size_t off = 0;
  auto alloc = [&](size_t n) { float* p = w + off; off += n; return p; };
  float* feat   = alloc((size_t)NT * DD);
  float* hvf    = alloc((size_t)NT * 2);
  float* feat2  = alloc((size_t)NT * 2);
  float* tables = alloc((size_t)3 * VV * DD);
  float* catb   = alloc((size_t)NA * DD);
  float* xcat   = alloc((size_t)NT * 2 * DD);
  float* r_t    = alloc((size_t)Et * DD);
  float* r_pt   = alloc((size_t)Ep * DD);
  float* r_a    = alloc((size_t)Ea * DD);
  float* efeat  = alloc((size_t)Emax * 4);
  float* sh     = w + off;   // shared scratch (fourier + attention)

  AttnWS aw;
  {
    size_t o2 = 0;
    auto a2 = [&](size_t n) { float* p = sh + o2; o2 += n; return p; };
    aw.xs  = a2((size_t)MS * DD);
    aw.xd  = a2((size_t)NT * DD);
    aw.q   = a2((size_t)NT * DD);
    aw.kb  = a2((size_t)MS * DD);
    aw.vb  = a2((size_t)MS * DD);
    aw.rr  = a2((size_t)Emax * DD);
    aw.ke  = a2((size_t)Emax * DD);
    aw.ve  = a2((size_t)Emax * DD);
    aw.sim = a2((size_t)Emax * NH);
    aw.mb  = a2((size_t)NT * NH);
    aw.den = a2((size_t)NT * NH);
    aw.cat2= a2((size_t)NT * 2 * DD);
    aw.gb  = a2((size_t)NT * DD);
    aw.sb  = a2((size_t)NT * DD);
    aw.hbig= a2((size_t)NT * 4 * DD);
  }

  const float* pos    = lf(d_in, IN_POS_A);
  const float* head   = lf(d_in, IN_HEAD_A);
  const float* ashape = lf(d_in, IN_AGENT_SHAPE);
  const int*   atype  = li(d_in, IN_AGENT_TYPE);
  const int*   tokidx = li(d_in, IN_TOKEN_IDX);
  const float* traj   = lf(d_in, IN_TRAJ);
  const float* x_pt   = lf(d_in, IN_X_PT);
  const float* pt_pos = lf(d_in, IN_PT_POS);
  const float* pt_or  = lf(d_in, IN_PT_ORIENT);
  const int* t_src = li(d_in, IN_T_SRC),   * t_dst = li(d_in, IN_T_DST);
  const int* p_src = li(d_in, IN_PT2A_SRC),* p_dst = li(d_in, IN_PT2A_DST);
  const int* a_src = li(d_in, IN_A2A_SRC), * a_dst = li(d_in, IN_A2A_DST);

  // ---------------- prologue ----------------
  k_tokfeat<<<(NT + 255) / 256, 256, 0, st>>>(pos, head, hvf, feat2);

  for (int t = 0; t < 3; ++t)
    mlp_apply(st, traj + (size_t)t * VV * 8, 8, VV, 8, DD, mw_tok, t,
              d_in, B_TOKEMB, tables + (size_t)t * VV * DD, DD, sh);
  {
    long n = (long)NT * DD;
    k_gather_tok<<<(unsigned)((n + 255) / 256), 256, 0, st>>>(tables, atype, tokidx, xcat);
  }
  {
    float* shp = sh + (size_t)NA * DD;
    mlp_apply(st, ashape, 3, NA, 3, DD, mw_shp, 0, d_in, B_SHAPE, shp, DD, sh);
    long n = (long)NA * DD;
    k_cat<<<(unsigned)((n + 255) / 256), 256, 0, st>>>(lf(d_in, IN_TYPE_EMB), atype, shp, catb);
  }
  fourier_apply(st, feat2, NT, 2, fw_xa, d_in, B_XA, catb, NS, xcat + DD, 2 * DD, sh);
  mlp_apply(st, xcat, 2 * DD, NT, 2 * DD, DD, mw_fus, 0, d_in, B_FUSION, feat, DD, sh);

  // ---------------- relation encodings ----------------
  k_efeat_t<<<(Et + 255) / 256, 256, 0, st>>>(pos, head, hvf, t_src, t_dst, efeat, Et);
  fourier_apply(st, efeat, Et, 4, fw_rt, d_in, B_RT, nullptr, 1, r_t, DD, sh);
  k_efeat_pt<<<(Ep + 255) / 256, 256, 0, st>>>(pt_pos, pt_or, pos, head, hvf,
                                               p_src, p_dst, efeat, Ep);
  fourier_apply(st, efeat, Ep, 3, fw_rp, d_in, B_RPT, nullptr, 1, r_pt, DD, sh);
  k_efeat_a<<<(Ea + 255) / 256, 256, 0, st>>>(pos, head, hvf, a_src, a_dst, efeat, Ea);
  fourier_apply(st, efeat, Ea, 3, fw_ra, d_in, B_RA, nullptr, 1, r_a, DD, sh);

  // ---------------- layer loop ----------------
  for (int l = 0; l < NLAYER; ++l) {
    attn_apply(st, d_in, B_TATTN,  l, pw_t,  feat, NT,  feat, r_t,  t_src, t_dst, Et, aw);
    attn_apply(st, d_in, B_PTATTN, l, pw_pt, x_pt, Ppt, feat, r_pt, p_src, p_dst, Ep, aw);
    attn_apply(st, d_in, B_AATTN,  l, pw_a,  feat, NT,  feat, r_a,  a_src, a_dst, Ea, aw);
  }

  // ---------------- head ----------------
  mlp_apply(st, feat, DD, NT, DD, VV, mw_head, 0, d_in, B_HEAD, (float*)d_out, VV, sh);
}
